// PointNetFeaturePropagation_50714973831837
// MI455X (gfx1250) — compile-verified, matrix operand order checked by reading, not run on hardware
//
#include <hip/hip_runtime.h>
#include <hip/hip_bf16.h>

typedef __attribute__((ext_vector_type(16))) __bf16 v16bf;
typedef __attribute__((ext_vector_type(8)))  float  v8f;

#define P_TOTAL 65536   // B*N = 16*4096 positions
#define N_PER_B 4096
#define S_PTS   1024
#define C_HALF  256     // C1 == C2 == 256
#define M_OUT   256     // output channels per layer

// ---------------------------------------------------------------------------
// WMMA fragment-layout helpers (CDNA5 16-bit A/B 16x32 layout, CDNA5 ISA):
//   lane = (row & 15) + 16*half,  half = (klocal>>3)&1
//   v    = ((klocal&16)?4:0) + ((klocal&7)>>1),  e = 2v + (klocal&1)
// Fragment f = rowTile*KT + kTile, element offset = lane*16 + e  (512 bf16/frag)
// ---------------------------------------------------------------------------
__device__ __forceinline__ void frag_decompose(int klocal, int& half, int& v) {
    half = (klocal >> 3) & 1;
    v    = ((klocal & 16) >> 2) + ((klocal & 7) >> 1);
}

// store a (k, k+1) bf16 pair for position p (k even) into fragment-layout buffer
__device__ __forceinline__ void store_bf16_pair(__bf16* X, int KT, int p, int k,
                                                float lo, float hi) {
    int half, v;
    frag_decompose(k & 31, half, v);
    int lane = (p & 15) + (half << 4);
    size_t idx = ((size_t)((p >> 4) * KT + (k >> 5))) * 512 + lane * 16 + 2 * v;
    union { __bf16 h[2]; unsigned u; } un;
    un.h[0] = (__bf16)lo; un.h[1] = (__bf16)hi;
    *(unsigned*)(X + idx) = un.u;   // idx even -> 4B aligned
}

// ---------------------------------------------------------------------------
// 0) zero the stats accumulators (4 arrays of 256 floats)
// ---------------------------------------------------------------------------
__global__ __launch_bounds__(256) void zero_stats_kernel(float* s) {
    int t = threadIdx.x;
    #pragma unroll
    for (int k = 0; k < 4; ++k) s[k * 256 + t] = 0.0f;
}

// ---------------------------------------------------------------------------
// 1) convert fp32 row-major weights [M,K] to bf16 A-matrix fragment layout
// ---------------------------------------------------------------------------
template<int K>
__global__ __launch_bounds__(256) void prep_weights_kernel(const float* __restrict__ w,
                                                           __bf16* __restrict__ wf) {
    const int KT = K / 32;
    int idx = blockIdx.x * 256 + threadIdx.x;     // m*K + k
    int m = idx / K, k = idx % K;
    int half, v;
    frag_decompose(k & 31, half, v);
    int lane = (m & 15) + (half << 4);
    int e = 2 * v + (k & 1);
    size_t a = ((size_t)((m >> 4) * KT + (k >> 5))) * 512 + lane * 16 + e;
    wf[a] = (__bf16)w[idx];
}

// ---------------------------------------------------------------------------
// 2) 3-NN inverse-distance interpolation + concat -> X bf16 in B-fragment
//    layout, K=512 (channels), P=65536 (positions). One thread per position.
// ---------------------------------------------------------------------------
__global__ __launch_bounds__(256) void interp_kernel(const float* __restrict__ xyz1,
                                                     const float* __restrict__ xyz2,
                                                     const float* __restrict__ points1,
                                                     const float* __restrict__ points2,
                                                     __bf16* __restrict__ Xfrag) {
    __shared__ float sx[S_PTS], sy[S_PTS], sz[S_PTS];
    const int b  = blockIdx.x >> 4;
    const int n  = (blockIdx.x & 15) * 256 + threadIdx.x;
    const int p  = b * N_PER_B + n;

    for (int s = threadIdx.x; s < S_PTS; s += 256) {
        sx[s] = xyz2[(b * 3 + 0) * S_PTS + s];
        sy[s] = xyz2[(b * 3 + 1) * S_PTS + s];
        sz[s] = xyz2[(b * 3 + 2) * S_PTS + s];
    }
    __syncthreads();

    const float qx = xyz1[(b * 3 + 0) * N_PER_B + n];
    const float qy = xyz1[(b * 3 + 1) * N_PER_B + n];
    const float qz = xyz1[(b * 3 + 2) * N_PER_B + n];

    float d0 = 3.0e38f, d1 = 3.0e38f, d2 = 3.0e38f;
    int   i0 = 0, i1 = 0, i2 = 0;
    for (int s = 0; s < S_PTS; ++s) {
        float dx = qx - sx[s], dy = qy - sy[s], dz = qz - sz[s];
        float dist = dx * dx + dy * dy + dz * dz;
        if (dist < d2) {
            if (dist < d1) {
                if (dist < d0) { d2 = d1; i2 = i1; d1 = d0; i1 = i0; d0 = dist; i0 = s; }
                else           { d2 = d1; i2 = i1; d1 = dist; i1 = s; }
            } else             { d2 = dist; i2 = s; }
        }
    }
    float r0 = 1.0f / (d0 + 1e-8f);
    float r1 = 1.0f / (d1 + 1e-8f);
    float r2 = 1.0f / (d2 + 1e-8f);
    float inv = 1.0f / (r0 + r1 + r2);
    float w0 = r0 * inv, w1 = r1 * inv, w2 = r2 * inv;

    const int KT = 512 / 32;   // 16 k-tiles
    const float* p1b = points1 + (size_t)b * C_HALF * N_PER_B + n;
    const float* p2b = points2 + (size_t)b * C_HALF * S_PTS;
    for (int c = 0; c < C_HALF; c += 2) {
        // channels 0..255 : points1 passthrough
        float a0 = p1b[(size_t)c * N_PER_B];
        float a1 = p1b[(size_t)(c + 1) * N_PER_B];
        store_bf16_pair(Xfrag, KT, p, c, a0, a1);
        // channels 256..511 : interpolated points2
        const float* rc0 = p2b + (size_t)c * S_PTS;
        const float* rc1 = p2b + (size_t)(c + 1) * S_PTS;
        float e0 = w0 * rc0[i0] + w1 * rc0[i1] + w2 * rc0[i2];
        float e1 = w0 * rc1[i0] + w1 * rc1[i1] + w2 * rc1[i2];
        store_bf16_pair(Xfrag, KT, p, C_HALF + c, e0, e1);
    }
}

// ---------------------------------------------------------------------------
// 3) bf16 WMMA GEMM  Y[256,65536] = A[256,K] * B[K,65536] + bias,
//    fused per-channel sum / sum-of-squares accumulation for BatchNorm.
//    Block tile 128x128, 8 waves, each wave 64x32 (4x2 wmma accumulators).
// ---------------------------------------------------------------------------
template<int KT>   // K = KT*32
__global__ __launch_bounds__(256) void gemm_bias_stats_kernel(
        const __bf16* __restrict__ Afrag, const __bf16* __restrict__ Bfrag,
        const float* __restrict__ bias, float* __restrict__ Y,
        float* __restrict__ sum, float* __restrict__ sumsq) {
    const int P = P_TOTAL;
    const int tid = threadIdx.x, wave = tid >> 5, lane = tid & 31;
    const int wm = wave >> 2, wn = wave & 3;
    const int bm = blockIdx.y * 128, bp = blockIdx.x * 128;
    const int mt0 = (bm + wm * 64) >> 4;   // 4 consecutive m-tiles
    const int pt0 = (bp + wn * 32) >> 4;   // 2 consecutive p-tiles

    v8f acc[4][2];
    #pragma unroll
    for (int i = 0; i < 4; ++i)
        #pragma unroll
        for (int j = 0; j < 2; ++j) acc[i][j] = v8f{};

    for (int kt = 0; kt < KT; ++kt) {
        const __bf16* ap = Afrag + ((size_t)(mt0 * KT + kt)) * 512 + lane * 16;
        const __bf16* bp_ = Bfrag + ((size_t)(pt0 * KT + kt)) * 512 + lane * 16;
        v16bf af[4], bf[2];
        #pragma unroll
        for (int i = 0; i < 4; ++i) af[i] = *(const v16bf*)(ap + (size_t)i * KT * 512);
        #pragma unroll
        for (int j = 0; j < 2; ++j) bf[j] = *(const v16bf*)(bp_ + (size_t)j * KT * 512);
        if (kt + 1 < KT) {                       // global_prefetch_b8 next k-step
            __builtin_prefetch(ap + 512, 0, 3);
            __builtin_prefetch(bp_ + 512, 0, 3);
        }
        #pragma unroll
        for (int i = 0; i < 4; ++i)
            #pragma unroll
            for (int j = 0; j < 2; ++j)
                acc[i][j] = __builtin_amdgcn_wmma_f32_16x16x32_bf16(
                    false, af[i], false, bf[j], (short)0, acc[i][j], false, false);
    }

    // epilogue: +bias, store fp32 Y, accumulate BN statistics
    const int halfsel = (lane >> 4) << 3;   // +8 channels in upper lane half
    #pragma unroll
    for (int i = 0; i < 4; ++i) {
        #pragma unroll
        for (int j = 0; j < 2; ++j) {
            int cb = bm + wm * 64 + i * 16 + halfsel;
            int pp = bp + wn * 32 + j * 16 + (lane & 15);
            #pragma unroll
            for (int r = 0; r < 8; ++r) {
                int c = cb + r;
                float vv = acc[i][j][r] + bias[c];
                Y[(size_t)c * P + pp] = vv;
                float s = vv, s2 = vv * vv;
                #pragma unroll
                for (int m = 1; m < 16; m <<= 1) {
                    s  += __shfl_xor(s,  m, 32);
                    s2 += __shfl_xor(s2, m, 32);
                }
                if ((lane & 15) == 0) {
                    atomicAdd(&sum[c],   s);
                    atomicAdd(&sumsq[c], s2);
                }
            }
        }
    }
}

// ---------------------------------------------------------------------------
// 4) fold BN stats into per-channel scale/shift:  y = x*scale + shift
// ---------------------------------------------------------------------------
__global__ __launch_bounds__(256) void finalize_stats_kernel(
        const float* __restrict__ sum, const float* __restrict__ sumsq,
        const float* __restrict__ g, const float* __restrict__ be,
        float* __restrict__ scale, float* __restrict__ shift) {
    int c = threadIdx.x;
    const float invN = 1.0f / (float)P_TOTAL;
    float mean = sum[c] * invN;
    float var  = sumsq[c] * invN - mean * mean;
    float is   = rsqrtf(var + 1e-5f);
    float sc   = g[c] * is;
    scale[c] = sc;
    shift[c] = be[c] - mean * sc;
}

// ---------------------------------------------------------------------------
// 5) BN+ReLU, re-emit as bf16 B-fragment layout (K=256) for the second GEMM
// ---------------------------------------------------------------------------
__global__ __launch_bounds__(256) void apply_bn_frag_kernel(
        const float* __restrict__ Y, const float* __restrict__ scale,
        const float* __restrict__ shift, __bf16* __restrict__ Zfrag) {
    const int P = P_TOTAL;
    int c = blockIdx.y * 2;
    int p = blockIdx.x * 256 + threadIdx.x;
    float z0 = fmaxf(Y[(size_t)c * P + p]       * scale[c]     + shift[c],     0.0f);
    float z1 = fmaxf(Y[(size_t)(c + 1) * P + p] * scale[c + 1] + shift[c + 1], 0.0f);
    store_bf16_pair(Zfrag, 256 / 32, p, c, z0, z1);
}

// ---------------------------------------------------------------------------
// 6) final BN+ReLU, write fp32 output in [B, 256, N] layout
// ---------------------------------------------------------------------------
__global__ __launch_bounds__(256) void apply_bn_out_kernel(
        const float* __restrict__ Y, const float* __restrict__ scale,
        const float* __restrict__ shift, float* __restrict__ out) {
    const int P = P_TOTAL;
    int c = blockIdx.y;
    int p = blockIdx.x * 256 + threadIdx.x;
    float v = fmaxf(Y[(size_t)c * P + p] * scale[c] + shift[c], 0.0f);
    int b = p >> 12, n = p & (N_PER_B - 1);
    out[(((size_t)b * M_OUT) + c) * N_PER_B + n] = v;
}

// ---------------------------------------------------------------------------
extern "C" void kernel_launch(void* const* d_in, const int* in_sizes, int n_in,
                              void* d_out, int out_size, void* d_ws, size_t ws_size,
                              hipStream_t stream) {
    const float* xyz1    = (const float*)d_in[0];
    const float* xyz2    = (const float*)d_in[1];
    const float* points1 = (const float*)d_in[2];
    const float* points2 = (const float*)d_in[3];
    const float* w0 = (const float*)d_in[4];
    const float* b0 = (const float*)d_in[5];
    const float* g0 = (const float*)d_in[6];
    const float* be0 = (const float*)d_in[7];
    const float* w1 = (const float*)d_in[8];
    const float* b1 = (const float*)d_in[9];
    const float* g1 = (const float*)d_in[10];
    const float* be1 = (const float*)d_in[11];

    char* ws = (char*)d_ws;
    const size_t OFF_X  = 0;                                    // X bf16 frag  [512,65536]
    const size_t OFF_W0 = OFF_X  + (size_t)P_TOTAL * 512 * 2;   // 67,108,864
    const size_t OFF_W1 = OFF_W0 + (size_t)256 * 512 * 2;
    const size_t OFF_Z  = OFF_W1 + (size_t)256 * 256 * 2;       // Z bf16 frag  [256,65536]
    const size_t OFF_Y  = OFF_Z  + (size_t)P_TOTAL * 256 * 2;   // Y fp32       [256,65536]
    const size_t OFF_S  = OFF_Y  + (size_t)256 * P_TOTAL * 4;   // stats

    __bf16* Xfrag = (__bf16*)(ws + OFF_X);
    __bf16* W0f   = (__bf16*)(ws + OFF_W0);
    __bf16* W1f   = (__bf16*)(ws + OFF_W1);
    __bf16* Zfrag = (__bf16*)(ws + OFF_Z);
    float*  Y     = (float*)(ws + OFF_Y);
    float*  stats = (float*)(ws + OFF_S);
    float *sum0 = stats,        *sq0 = stats + 256;
    float *sum1 = stats + 512,  *sq1 = stats + 768;
    float *sc0  = stats + 1024, *sh0 = stats + 1280;
    float *sc1  = stats + 1536, *sh1 = stats + 1792;

    zero_stats_kernel<<<1, 256, 0, stream>>>(stats);
    prep_weights_kernel<512><<<(256 * 512) / 256, 256, 0, stream>>>(w0, W0f);
    prep_weights_kernel<256><<<(256 * 256) / 256, 256, 0, stream>>>(w1, W1f);
    interp_kernel<<<256, 256, 0, stream>>>(xyz1, xyz2, points1, points2, Xfrag);

    gemm_bias_stats_kernel<16><<<dim3(512, 2), 256, 0, stream>>>(W0f, Xfrag, b0, Y, sum0, sq0);
    finalize_stats_kernel<<<1, 256, 0, stream>>>(sum0, sq0, g0, be0, sc0, sh0);
    apply_bn_frag_kernel<<<dim3(256, 128), 256, 0, stream>>>(Y, sc0, sh0, Zfrag);

    gemm_bias_stats_kernel<8><<<dim3(512, 2), 256, 0, stream>>>(W1f, Zfrag, b1, Y, sum1, sq1);
    finalize_stats_kernel<<<1, 256, 0, stream>>>(sum1, sq1, g1, be1, sc1, sh1);
    apply_bn_out_kernel<<<dim3(256, 256), 256, 0, stream>>>(Y, sc1, sh1, (float*)d_out);
}
